// STEM_48541720379867
// MI455X (gfx1250) — compile-verified, matrix operand order checked by reading, not run on hardware
//
#include <hip/hip_runtime.h>
#include <hip/hip_bf16.h>

// ---------------------------------------------------------------------------
// Problem constants (from reference)
// ---------------------------------------------------------------------------
#define BATCH 16384
#define TNUM  3
#define GRP   4            // expert groups
#define NEXP  4            // experts per group
#define NXP   16           // GRP*NEXP
#define DIM   512
#define HID   512
#define EMB   256
#define GH    256
#define TH    128
#define TO    64

typedef __attribute__((ext_vector_type(8)))  float          v8f;
typedef __attribute__((ext_vector_type(8)))  unsigned short v8u;
typedef __attribute__((ext_vector_type(16))) unsigned short v16u;
typedef __attribute__((ext_vector_type(16))) __bf16         v16bf;
typedef __attribute__((ext_vector_type(4)))  int            v4i;

// CDNA5 async global->LDS path (ASYNCcnt), with safe fallback.
#if defined(__AMDGCN__) && __has_builtin(__builtin_amdgcn_global_load_async_to_lds_b128)
#define USE_ASYNC_LDS 1
#else
#define USE_ASYNC_LDS 0
#endif

#if USE_ASYNC_LDS
typedef __attribute__((address_space(1))) v4i* gv4i_p;   // global (AS1)
typedef __attribute__((address_space(3))) v4i* lv4i_p;   // LDS    (AS3)
__device__ __forceinline__ void async_cp16(const unsigned short* g, unsigned short* l) {
    __builtin_amdgcn_global_load_async_to_lds_b128((gv4i_p)g, (lv4i_p)l, 0, 0);
}
__device__ __forceinline__ void wait_async0() {
#if __has_builtin(__builtin_amdgcn_s_wait_asynccnt)
    __builtin_amdgcn_s_wait_asynccnt(0);
#else
    asm volatile("s_wait_asynccnt 0x0" ::: "memory");
#endif
}
#endif

// float -> bf16 (round to nearest even), bf16 -> float
__device__ __forceinline__ unsigned short f2bf(float f) {
    unsigned u = __float_as_uint(f);
    u = (u + 0x7FFFu + ((u >> 16) & 1u)) >> 16;
    return (unsigned short)u;
}
__device__ __forceinline__ float bf2f(unsigned short h) {
    return __uint_as_float(((unsigned)h) << 16);
}

__device__ __forceinline__ v8f vzero8() {
    v8f z = {0.f, 0.f, 0.f, 0.f, 0.f, 0.f, 0.f, 0.f};
    return z;
}

// WMMA bf16: D = A(16x32) * B(32x16) + C  (f32 accum)
__device__ __forceinline__ v8f wmma_bf16(v16u a, v16u b, v8f c) {
    return __builtin_amdgcn_wmma_f32_16x16x32_bf16(
        false, __builtin_bit_cast(v16bf, a),
        false, __builtin_bit_cast(v16bf, b),
        (short)0, c, false, false);
}

// Load one 16x32 bf16 A/B fragment half-pair for this lane.
// Caller passes base = &M[row * stride + k0 + koff]  (koff = (lane>>4)*8)
// Fragment per-lane: elems 0..7 = K(koff..koff+7), 8..15 = K(koff+16..koff+23)
__device__ __forceinline__ v16u load_frag(const unsigned short* base) {
    v8u lo = *(const v8u*)(base);
    v8u hi = *(const v8u*)(base + 16);
    v16u t;
#pragma unroll
    for (int e = 0; e < 8; ++e) { t[e] = lo[e]; t[8 + e] = hi[e]; }
    return t;
}

// ---------------------------------------------------------------------------
// Prep: inputs fp32 [B,G,D] -> bf16 same layout; gate input [T,B,D] bf16
// ---------------------------------------------------------------------------
__global__ __launch_bounds__(256)
void k_prep_inputs(const float* __restrict__ x,
                   unsigned short* __restrict__ xbf,
                   unsigned short* __restrict__ gin) {
    size_t idx = (size_t)blockIdx.x * 256 + threadIdx.x;   // over B*D exactly
    int b = (int)(idx >> 9);         // /512
    int d = (int)(idx & 511);
    const float* p = x + (size_t)b * (GRP * DIM) + d;
    float x0 = p[0], x1 = p[DIM], x2 = p[2 * DIM], x3 = p[3 * DIM];
    unsigned short* q = xbf + (size_t)b * (GRP * DIM) + d;
    q[0]       = f2bf(x0);
    q[DIM]     = f2bf(x1);
    q[2 * DIM] = f2bf(x2);
    q[3 * DIM] = f2bf(x3);
    size_t bd = (size_t)b * DIM + d;
    gin[bd]                           = f2bf(x1 + x0);
    gin[(size_t)BATCH * DIM + bd]     = f2bf(x2 + x0);
    gin[2 * (size_t)BATCH * DIM + bd] = f2bf(x3 + x0);
}

// ---------------------------------------------------------------------------
// Weight transpose+convert: in [Z,K,N] fp32 -> out [Z,N,K] bf16
// ---------------------------------------------------------------------------
__global__ __launch_bounds__(256)
void k_wtranspose(const float* __restrict__ in, unsigned short* __restrict__ out,
                  int Kd, int Nd) {
    size_t total = (size_t)Kd * Nd;
    size_t idx = (size_t)blockIdx.x * 256 + threadIdx.x;
    if (idx >= total) return;
    int n = (int)(idx / Kd);
    int k = (int)(idx % Kd);
    int z = blockIdx.y;
    out[(size_t)z * total + idx] = f2bf(in[(size_t)z * total + (size_t)k * Nd + n]);
}

// ---------------------------------------------------------------------------
// Generic batched WMMA GEMM:  C[z] = act(A[z] * Wt[z]^T + bias[z])
//   A  : bf16, rows M, row-stride lda, batch offset (z>>aShift)*aBatch
//   Wt : bf16 [N,K] rows (pre-transposed weights), batch z*wBatch
//   C  : bf16, row-stride ldc, batch z*cBatch
// Block tile 256x128, 8 waves (4Mx2N), wave tile 64x64 (16 WMMAs/K-tile),
// K-tile 32 staged in double-buffered LDS via async global->LDS loads.
// Requires M%256==0, N%128==0, K%32==0.
// ---------------------------------------------------------------------------
#define LSTR 40   // LDS row pitch in bf16 elems (80B): conflict-free b128 reads

template <bool RELU>
__global__ __launch_bounds__(256)
void k_gemm(const unsigned short* __restrict__ A, long long aBatch, int aShift, int lda,
            const unsigned short* __restrict__ Wt, long long wBatch,
            const float* __restrict__ bias, int bBatch,
            unsigned short* __restrict__ C, long long cBatch, int ldc,
            int K) {
    __shared__ unsigned short At[2][256 * LSTR];
    __shared__ unsigned short Bt[2][128 * LSTR];

    const int tid  = threadIdx.x;
    const int lane = tid & 31;
    const int wave = tid >> 5;
    const int wm   = wave >> 1;     // 0..3 -> 64-row slab
    const int wn   = wave & 1;      // 0..1 -> 64-col slab
    const int ln   = lane & 15;
    const int lh   = lane >> 4;
    const int koff = lh * 8;

    const int z  = blockIdx.z;
    const int m0 = blockIdx.x * 256;
    const int n0 = blockIdx.y * 128;

    const unsigned short* Ab = A  + (size_t)(z >> aShift) * (size_t)aBatch;
    const unsigned short* Wb = Wt + (size_t)z * (size_t)wBatch;

    v8f acc[4][4];
#pragma unroll
    for (int i = 0; i < 4; ++i)
#pragma unroll
        for (int j = 0; j < 4; ++j) acc[i][j] = vzero8();

    // ---- K-tile staging: 256x32 A chunk + 128x32 B chunk ----------------
    auto stage = [&](int buf, int k0) {
#pragma unroll
        for (int t = 0; t < 4; ++t) {            // A: 1024 16B-chunks
            int c = tid + t * 256;
            int r = c >> 2, col = (c & 3) << 3;
            const unsigned short* g = Ab + (size_t)(m0 + r) * lda + (k0 + col);
            unsigned short* l = &At[buf][r * LSTR + col];
#if USE_ASYNC_LDS
            async_cp16(g, l);
#else
            *(v8u*)l = *(const v8u*)g;
#endif
        }
#pragma unroll
        for (int t = 0; t < 2; ++t) {            // B: 512 16B-chunks
            int c = tid + t * 256;
            int r = c >> 2, col = (c & 3) << 3;
            const unsigned short* g = Wb + (size_t)(n0 + r) * K + (k0 + col);
            unsigned short* l = &Bt[buf][r * LSTR + col];
#if USE_ASYNC_LDS
            async_cp16(g, l);
#else
            *(v8u*)l = *(const v8u*)g;
#endif
        }
    };

    const int nk = K >> 5;
    stage(0, 0);
    for (int kt = 0; kt < nk; ++kt) {
        const int cur = kt & 1;
#if USE_ASYNC_LDS
        wait_async0();                 // my async loads for buffer `cur` done
#endif
        __syncthreads();               // everyone's are done -> tile readable
        if (kt + 1 < nk) stage(cur ^ 1, (kt + 1) * 32);   // prefetch next tile

        v16u af[4], bfr[4];
#pragma unroll
        for (int i = 0; i < 4; ++i) {
            int r = wm * 64 + i * 16 + ln;
            af[i] = load_frag(&At[cur][r * LSTR + koff]);
        }
#pragma unroll
        for (int j = 0; j < 4; ++j) {
            int r = wn * 64 + j * 16 + ln;
            bfr[j] = load_frag(&Bt[cur][r * LSTR + koff]);
        }
#pragma unroll
        for (int i = 0; i < 4; ++i)
#pragma unroll
            for (int j = 0; j < 4; ++j)
                acc[i][j] = wmma_bf16(af[i], bfr[j], acc[i][j]);
    }

    // ---- epilogue: bias (+relu), bf16 store -----------------------------
#pragma unroll
    for (int j = 0; j < 4; ++j) {
        int col = n0 + wn * 64 + j * 16 + ln;
        float bv = bias ? bias[(size_t)z * bBatch + col] : 0.f;
#pragma unroll
        for (int i = 0; i < 4; ++i) {
            int rowb = m0 + wm * 64 + i * 16 + lh * 8;
#pragma unroll
            for (int v = 0; v < 8; ++v) {
                float x = acc[i][j][v] + bv;
                if (RELU) x = fmaxf(x, 0.f);
                C[(size_t)z * cBatch + (size_t)(rowb + v) * ldc + col] = f2bf(x);
            }
        }
    }
}

// ---------------------------------------------------------------------------
// Gate layer 2 (GH=256 -> 16) + softmax. One wave per 16 rows per task.
// C layout: lane = n + 16*(m>=8), vgpr v = m&7  -> all 16 logits of a row
// live across a 16-lane half; reduce with shfl_xor(1..8).
// ---------------------------------------------------------------------------
__global__ __launch_bounds__(32)
void k_gate2_softmax(const unsigned short* __restrict__ gh,   // [T,B,GH] bf16
                     const unsigned short* __restrict__ w2t,  // [T,16,GH] bf16
                     const float* __restrict__ b2,            // [T,16]
                     float* __restrict__ gw) {                // [B,T,16]
    const int lane = threadIdx.x;
    const int ln = lane & 15, lh = lane >> 4, koff = lh * 8;
    const int t  = blockIdx.y;
    const int m0 = blockIdx.x * 16;

    const unsigned short* Ab = gh + ((size_t)t * BATCH + m0) * GH;
    const unsigned short* Wb = w2t + (size_t)t * 16 * GH;

    v8f acc = vzero8();
    for (int k0 = 0; k0 < GH; k0 += 32) {
        v16u a = load_frag(Ab + (size_t)ln * GH + k0 + koff);
        v16u b = load_frag(Wb + (size_t)ln * GH + k0 + koff);
        acc = wmma_bf16(a, b, acc);
    }
    float bv = b2[t * 16 + ln];
#pragma unroll
    for (int v = 0; v < 8; ++v) {
        float x = acc[v] + bv;
        float mx = x;
#pragma unroll
        for (int s = 8; s >= 1; s >>= 1) mx = fmaxf(mx, __shfl_xor(mx, s, 32));
        float e = __expf(x - mx);
        float sum = e;
#pragma unroll
        for (int s = 8; s >= 1; s >>= 1) sum += __shfl_xor(sum, s, 32);
        int row = m0 + lh * 8 + v;
        gw[((size_t)row * TNUM + t) * 16 + ln] = e / sum;
    }
}

// ---------------------------------------------------------------------------
// Aggregation: agg[t,b,:] = sum_k gw[b,t,k] * emb[b,k,:]   (k=16, e=256)
// ---------------------------------------------------------------------------
__global__ __launch_bounds__(256)
void k_aggregate(const float* __restrict__ gw,            // [B,T,16]
                 const unsigned short* __restrict__ emb,  // [B,16,EMB] bf16
                 unsigned short* __restrict__ agg) {      // [T,B,EMB] bf16
    const int b = blockIdx.x, t = blockIdx.y, e = threadIdx.x;
    __shared__ float w[NXP];
    if (threadIdx.x < NXP) w[threadIdx.x] = gw[((size_t)b * TNUM + t) * NXP + threadIdx.x];
    __syncthreads();
    const unsigned short* p = emb + (size_t)b * (NXP * EMB) + e;
    float s = 0.f;
#pragma unroll
    for (int k = 0; k < NXP; ++k) s += w[k] * bf2f(p[k * EMB]);
    agg[((size_t)t * BATCH + b) * EMB + e] = f2bf(s);
}

// ---------------------------------------------------------------------------
// Tower layer 2 (TH=128 -> TO=64) + final output [B, T*TO] fp32.
// One wave per 16 rows per task; 4 WMMA column tiles.
// ---------------------------------------------------------------------------
__global__ __launch_bounds__(32)
void k_tower2(const unsigned short* __restrict__ th,    // [T,B,TH] bf16
              const unsigned short* __restrict__ w2t,   // [T,TO,TH] bf16
              const float* __restrict__ b2,             // [T,TO]
              float* __restrict__ out) {                // [B, T*TO]
    const int lane = threadIdx.x;
    const int ln = lane & 15, lh = lane >> 4, koff = lh * 8;
    const int t  = blockIdx.y;
    const int m0 = blockIdx.x * 16;

    const unsigned short* Ab = th + ((size_t)t * BATCH + m0) * TH;
    const unsigned short* Wb = w2t + (size_t)t * TO * TH;

    v8f acc[4];
#pragma unroll
    for (int j = 0; j < 4; ++j) acc[j] = vzero8();

    for (int k0 = 0; k0 < TH; k0 += 32) {
        v16u a = load_frag(Ab + (size_t)ln * TH + k0 + koff);
#pragma unroll
        for (int j = 0; j < 4; ++j) {
            v16u b = load_frag(Wb + (size_t)(j * 16 + ln) * TH + k0 + koff);
            acc[j] = wmma_bf16(a, b, acc[j]);
        }
    }
#pragma unroll
    for (int j = 0; j < 4; ++j) {
        int col = j * 16 + ln;
        float bv = b2[t * TO + col];
#pragma unroll
        for (int v = 0; v < 8; ++v) {
            int row = m0 + lh * 8 + v;
            out[(size_t)row * (TNUM * TO) + t * TO + col] = acc[j][v] + bv;
        }
    }
}

// ---------------------------------------------------------------------------
// Host launcher
// ---------------------------------------------------------------------------
extern "C" void kernel_launch(void* const* d_in, const int* in_sizes, int n_in,
                              void* d_out, int out_size, void* d_ws, size_t ws_size,
                              hipStream_t stream) {
    const float* inputs    = (const float*)d_in[0];
    const float* expert_W1 = (const float*)d_in[1];
    const float* expert_b1 = (const float*)d_in[2];
    const float* expert_W2 = (const float*)d_in[3];
    const float* expert_b2 = (const float*)d_in[4];
    const float* gate_W1   = (const float*)d_in[5];
    const float* gate_b1   = (const float*)d_in[6];
    const float* gate_W2   = (const float*)d_in[7];
    const float* gate_b2   = (const float*)d_in[8];
    const float* tower_W1  = (const float*)d_in[9];
    const float* tower_b1  = (const float*)d_in[10];
    const float* tower_W2  = (const float*)d_in[11];
    const float* tower_b2  = (const float*)d_in[12];
    float* out = (float*)d_out;

    // ---- workspace carve-up (bf16 unless noted) -------------------------
    char* w = (char*)d_ws;
    size_t off = 0;
    auto take = [&](size_t bytes) {
        char* p = w + off;
        off += (bytes + 255) & ~(size_t)255;
        return p;
    };
    unsigned short* Xbf  = (unsigned short*)take((size_t)BATCH * GRP * DIM * 2);
    unsigned short* gin  = (unsigned short*)take((size_t)TNUM * BATCH * DIM * 2);
    unsigned short* W1t  = (unsigned short*)take((size_t)NXP * HID * DIM * 2);
    unsigned short* W2t  = (unsigned short*)take((size_t)NXP * EMB * HID * 2);
    unsigned short* gW1t = (unsigned short*)take((size_t)TNUM * GH * DIM * 2);
    unsigned short* gW2t = (unsigned short*)take((size_t)TNUM * 16 * GH * 2);
    unsigned short* tW1t = (unsigned short*)take((size_t)TNUM * TH * EMB * 2);
    unsigned short* tW2t = (unsigned short*)take((size_t)TNUM * TO * TH * 2);
    unsigned short* H1   = (unsigned short*)take((size_t)NXP * BATCH * HID * 2);
    unsigned short* embb = (unsigned short*)take((size_t)BATCH * NXP * EMB * 2);
    unsigned short* ghb  = (unsigned short*)take((size_t)TNUM * BATCH * GH * 2);
    float*          gwb  = (float*)         take((size_t)BATCH * TNUM * 16 * 4);
    unsigned short* aggb = (unsigned short*)take((size_t)TNUM * BATCH * EMB * 2);
    unsigned short* thb  = (unsigned short*)take((size_t)TNUM * BATCH * TH * 2);
    (void)ws_size; (void)in_sizes; (void)n_in; (void)out_size;

    // ---- 1. convert inputs / gate inputs --------------------------------
    k_prep_inputs<<<(BATCH * DIM) / 256, 256, 0, stream>>>(inputs, Xbf, gin);

    // ---- 2. transpose+convert weights to [Z,N,K] bf16 -------------------
    auto wt = [&](const float* in, unsigned short* o, int Z, int K, int N) {
        dim3 g((unsigned)(((size_t)K * N + 255) / 256), Z);
        k_wtranspose<<<g, 256, 0, stream>>>(in, o, K, N);
    };
    wt(expert_W1, W1t,  NXP, DIM, HID);   // [16,512,512] -> [16,512,512]^T
    wt(expert_W2, W2t,  NXP, HID, EMB);   // [16,512,256] -> [16,256,512]
    wt(gate_W1,   gW1t, TNUM, DIM, GH);   // [3,512,256]  -> [3,256,512]
    wt(gate_W2,   gW2t, TNUM, GH, 16);    // [3,256,16]   -> [3,16,256]
    wt(tower_W1,  tW1t, TNUM, EMB, TH);   // [3,256,128]  -> [3,128,256]
    wt(tower_W2,  tW2t, TNUM, TH, TO);    // [3,128,64]   -> [3,64,128]

    // ---- 3. expert layer 1: H1[z] = relu(X[g] @ W1[z]) ------------------
    k_gemm<true><<<dim3(BATCH / 256, HID / 128, NXP), 256, 0, stream>>>(
        Xbf, (long long)DIM, /*aShift=*/2, GRP * DIM,
        W1t, (long long)HID * DIM,
        expert_b1, HID,
        H1, (long long)BATCH * HID, HID, DIM);

    // ---- 4. expert layer 2: emb[:,z,:] = H1[z] @ W2[z] ------------------
    k_gemm<false><<<dim3(BATCH / 256, EMB / 128, NXP), 256, 0, stream>>>(
        H1, (long long)BATCH * HID, 0, HID,
        W2t, (long long)EMB * HID,
        expert_b2, EMB,
        embb, (long long)EMB, NXP * EMB, HID);

    // ---- 5. gate layer 1: gh[t] = relu(gin[t] @ gW1[t]) -----------------
    k_gemm<true><<<dim3(BATCH / 256, GH / 128, TNUM), 256, 0, stream>>>(
        gin, (long long)BATCH * DIM, 0, DIM,
        gW1t, (long long)GH * DIM,
        gate_b1, GH,
        ghb, (long long)BATCH * GH, GH, DIM);

    // ---- 6. gate layer 2 + softmax --------------------------------------
    k_gate2_softmax<<<dim3(BATCH / 16, TNUM), 32, 0, stream>>>(ghb, gW2t, gate_b2, gwb);

    // ---- 7. gated aggregation over 16 experts ---------------------------
    k_aggregate<<<dim3(BATCH, TNUM), 256, 0, stream>>>(gwb, embb, aggb);

    // ---- 8. tower layer 1: th[t] = relu(agg[t] @ tW1[t]) ----------------
    k_gemm<true><<<dim3(BATCH / 256, TH / 128, TNUM), 256, 0, stream>>>(
        aggb, (long long)BATCH * EMB, 0, EMB,
        tW1t, (long long)TH * EMB,
        tower_b1, TH,
        thb, (long long)BATCH * TH, TH, EMB);

    // ---- 9. tower layer 2 -> final output -------------------------------
    k_tower2<<<dim3(BATCH / 16, TNUM), 32, 0, stream>>>(thb, tW2t, tower_b2, out);
}
